// ModifiedDoubleStreamBlock_10883447128466
// MI455X (gfx1250) — matrix operand order repaired and checked by reading, not compiled, not run on hardware
//
#include <hip/hip_runtime.h>

// ---------------------------------------------------------------------------
// CDNA5 (gfx1250) types / helpers
// ---------------------------------------------------------------------------
typedef __attribute__((ext_vector_type(16))) __bf16 bf16x16;
typedef __attribute__((ext_vector_type(8)))  float  f32x8;

union Frag { bf16x16 v; unsigned u[8]; };

__device__ __forceinline__ unsigned pack2_bf16(float a, float b) {
  union { float f; unsigned u; } x, y;
  x.f = a; y.f = b;
  unsigned lo = (x.u + 0x7FFFu + ((x.u >> 16) & 1u)) >> 16;
  unsigned hi = (y.u + 0x7FFFu + ((y.u >> 16) & 1u)) >> 16;
  return (hi << 16) | (lo & 0xFFFFu);
}

__device__ __forceinline__ __bf16 to_bf16(float a) {
  union { float f; unsigned u; } x; x.f = a;
  union { unsigned short s; __bf16 b; } r;
  r.s = (unsigned short)((x.u + 0x7FFFu + ((x.u >> 16) & 1u)) >> 16);
  return r.b;
}

__device__ __forceinline__ f32x8 wmma_bf16(bf16x16 a, bf16x16 b, f32x8 c) {
  // D = A(16x32 bf16) * B(32x16 bf16) + C(16x16 f32)
  return __builtin_amdgcn_wmma_f32_16x16x32_bf16(false, a, false, b, (short)0, c, false, false);
}

__device__ __forceinline__ float gelu_tanh(float x) {
  float x3 = x * x * x;
  return 0.5f * x * (1.f + tanhf(0.7978845608028654f * (x + 0.044715f * x3)));
}

// LDS byte-offset of a __shared__ object: low 32 bits of the generic pointer
// (LDS aperture: addr[63:32]=aperture tag, addr[31:0]=offset).
__device__ __forceinline__ unsigned lds_offset_of(const void* p) {
  return (unsigned)(size_t)p;
}

// ---------------------------------------------------------------------------
// Tiled WMMA GEMM (fp32 A and B, converted to bf16 on the fly):
//   C[M,N] = act(A[M,K] @ B + bias) (optionally gated residual)
//   BTRANS=0: B is [K,N] row-major.  BTRANS=1: B is [N,K] row-major
//   ACT: 0=none, 1=tanh-gelu ; GATED: 1 -> C = resid + gate[n] * v
// Tile: BM=128, BN=64, BK=32. 256 threads = 8 waves (4 along M x 2 along N).
// ---------------------------------------------------------------------------
template <int BTRANS, int ACT, int GATED>
__global__ __launch_bounds__(256) void gemm_kernel(
    const float* __restrict__ A, const float* __restrict__ B,
    const float* __restrict__ bias, const float* __restrict__ gate,
    const float* __restrict__ resid, float* __restrict__ C,
    int M, int N, int K, float out_scale)
{
  __shared__ unsigned aS[128][17];   // [m][kpair]  (bf16 pairs)
  __shared__ unsigned bS[64][17];    // [n][kpair]

  const int tid  = threadIdx.x;
  const int lane = tid & 31;
  const int wid  = tid >> 5;
  const int wm   = wid & 3;
  const int wn   = wid >> 2;
  const int ml   = lane & 15;
  const int hh   = lane >> 4;

  const int m0 = blockIdx.y * 128;
  const int n0 = blockIdx.x * 64;

  f32x8 acc[2][2];
#pragma unroll
  for (int mi = 0; mi < 2; ++mi)
#pragma unroll
    for (int ni = 0; ni < 2; ++ni)
#pragma unroll
      for (int e = 0; e < 8; ++e) acc[mi][ni][e] = 0.f;

  for (int k0 = 0; k0 < K; k0 += 32) {
#pragma unroll
    for (int i = 0; i < 8; ++i) {
      int p  = tid + i * 256;
      int r  = p >> 4;
      int kp = p & 15;
      int gm = m0 + r;
      float2 va = make_float2(0.f, 0.f);
      if (gm < M) va = *reinterpret_cast<const float2*>(A + (size_t)gm * K + k0 + 2 * kp);
      aS[r][kp] = pack2_bf16(va.x, va.y);
    }
    if (BTRANS) {
#pragma unroll
      for (int i = 0; i < 4; ++i) {
        int p  = tid + i * 256;
        int kp = p & 15;
        int n  = p >> 4;
        int gn = n0 + n;
        float2 vb = make_float2(0.f, 0.f);
        if (gn < N) vb = *reinterpret_cast<const float2*>(B + (size_t)gn * K + k0 + 2 * kp);
        bS[n][kp] = pack2_bf16(vb.x, vb.y);
      }
    } else {
#pragma unroll
      for (int i = 0; i < 4; ++i) {
        int p  = tid + i * 256;
        int n  = p & 63;
        int kp = p >> 6;
        int gn = n0 + n;
        int gk = k0 + 2 * kp;
        float f0 = 0.f, f1 = 0.f;
        if (gn < N) {
          f0 = B[(size_t)gk * N + gn];
          f1 = B[(size_t)(gk + 1) * N + gn];
        }
        bS[n][kp] = pack2_bf16(f0, f1);
      }
    }
    __syncthreads();

    Frag fa[2], fb[2];
#pragma unroll
    for (int mi = 0; mi < 2; ++mi) {
      int row = wm * 32 + mi * 16 + ml;
#pragma unroll
      for (int j = 0; j < 4; ++j) {
        fa[mi].u[j]     = aS[row][hh * 4 + j];
        fa[mi].u[4 + j] = aS[row][8 + hh * 4 + j];
      }
    }
#pragma unroll
    for (int ni = 0; ni < 2; ++ni) {
      int col = wn * 32 + ni * 16 + ml;
#pragma unroll
      for (int j = 0; j < 8; ++j)
        fb[ni].u[j] = bS[col][(j >> 2) * 8 + hh * 4 + (j & 3)];
    }
#pragma unroll
    for (int mi = 0; mi < 2; ++mi)
#pragma unroll
      for (int ni = 0; ni < 2; ++ni)
        acc[mi][ni] = wmma_bf16(fa[mi].v, fb[ni].v, acc[mi][ni]);
    __syncthreads();
  }

#pragma unroll
  for (int mi = 0; mi < 2; ++mi)
#pragma unroll
    for (int ni = 0; ni < 2; ++ni) {
      int gn = n0 + wn * 32 + ni * 16 + ml;
#pragma unroll
      for (int r = 0; r < 8; ++r) {
        int gm = m0 + wm * 32 + mi * 16 + hh * 8 + r;
        if (gm < M && gn < N) {
          float v = acc[mi][ni][r] * out_scale;
          if (bias) v += bias[gn];
          if (ACT == 1) v = gelu_tanh(v);
          if (GATED) v = resid[(size_t)gm * N + gn] + gate[gn] * v;
          C[(size_t)gm * N + gn] = v;
        }
      }
    }
}

// ---------------------------------------------------------------------------
// Weight pre-pass: W[K,N] fp32 -> Wt[N,K] bf16 (tiled transpose via LDS).
// K, N must be multiples of 32.
// ---------------------------------------------------------------------------
__global__ __launch_bounds__(256) void convT_kernel(
    const float* __restrict__ W, __bf16* __restrict__ Wt, int K, int N)
{
  __shared__ float t[32][33];
  const int kb = blockIdx.y * 32, nb = blockIdx.x * 32;
  const int tx = threadIdx.x & 31, ty = threadIdx.x >> 5;
#pragma unroll
  for (int i = 0; i < 4; ++i) {
    int k = ty + i * 8;
    t[k][tx] = W[(size_t)(kb + k) * N + nb + tx];
  }
  __syncthreads();
#pragma unroll
  for (int i = 0; i < 4; ++i) {
    int n = ty + i * 8;
    Wt[(size_t)(nb + n) * K + kb + tx] = to_bf16(t[tx][n]);
  }
}

// ---------------------------------------------------------------------------
// WMMA GEMM with pre-transposed bf16 weights, B tile fetched with CDNA5
// async global->LDS DMA (GLOBAL_LOAD_ASYNC_TO_LDS_B128, ASYNCcnt).
//   C[M,N] = act(A[M,K] @ Bt^T + bias) ; Bt is [N,K] bf16.
// Requires N % 64 == 0 and K % 32 == 0 (true for all call sites).
// bS rows are 80 bytes (16 data uints + 4 pad) so each 16B chunk is aligned.
// ---------------------------------------------------------------------------
template <int ACT, int GATED>
__global__ __launch_bounds__(256) void gemm_bf16t_kernel(
    const float* __restrict__ A, const __bf16* __restrict__ Bt,
    const float* __restrict__ bias, const float* __restrict__ gate,
    const float* __restrict__ resid, float* __restrict__ C,
    int M, int N, int K, float out_scale)
{
  __shared__ unsigned aS[128][17];
  __shared__ unsigned bS[64][20];   // pitch 80B -> every 16B chunk aligned

  const int tid  = threadIdx.x;
  const int lane = tid & 31;
  const int wid  = tid >> 5;
  const int wm   = wid & 3;
  const int wn   = wid >> 2;
  const int ml   = lane & 15;
  const int hh   = lane >> 4;

  const int m0 = blockIdx.y * 128;
  const int n0 = blockIdx.x * 64;

  // per-thread B chunk: row bn (0..63), 16B chunk bc (0..3)
  const int bn = tid >> 2;
  const int bc = tid & 3;
  const unsigned ldsB = lds_offset_of(&bS[bn][bc * 4]);

  f32x8 acc[2][2];
#pragma unroll
  for (int mi = 0; mi < 2; ++mi)
#pragma unroll
    for (int ni = 0; ni < 2; ++ni)
#pragma unroll
      for (int e = 0; e < 8; ++e) acc[mi][ni][e] = 0.f;

  for (int k0 = 0; k0 < K; k0 += 32) {
    // async DMA: 64B row chunks of the bf16 [N,K] weight into LDS
    {
      unsigned long long ga =
          (unsigned long long)(Bt + (size_t)(n0 + bn) * K + k0 + bc * 8);
      asm volatile("global_load_async_to_lds_b128 %0, %1, off"
                   :: "v"(ldsB), "v"(ga) : "memory");
    }
    // A tile through VGPRs (fp32 -> bf16 conversion)
#pragma unroll
    for (int i = 0; i < 8; ++i) {
      int p  = tid + i * 256;
      int r  = p >> 4;
      int kp = p & 15;
      int gm = m0 + r;
      float2 va = make_float2(0.f, 0.f);
      if (gm < M) va = *reinterpret_cast<const float2*>(A + (size_t)gm * K + k0 + 2 * kp);
      aS[r][kp] = pack2_bf16(va.x, va.y);
    }
    asm volatile("s_wait_asynccnt 0" ::: "memory");
    __syncthreads();

    Frag fa[2], fb[2];
#pragma unroll
    for (int mi = 0; mi < 2; ++mi) {
      int row = wm * 32 + mi * 16 + ml;
#pragma unroll
      for (int j = 0; j < 4; ++j) {
        fa[mi].u[j]     = aS[row][hh * 4 + j];
        fa[mi].u[4 + j] = aS[row][8 + hh * 4 + j];
      }
    }
#pragma unroll
    for (int ni = 0; ni < 2; ++ni) {
      int col = wn * 32 + ni * 16 + ml;
#pragma unroll
      for (int j = 0; j < 8; ++j)
        fb[ni].u[j] = bS[col][(j >> 2) * 8 + hh * 4 + (j & 3)];
    }
#pragma unroll
    for (int mi = 0; mi < 2; ++mi)
#pragma unroll
      for (int ni = 0; ni < 2; ++ni)
        acc[mi][ni] = wmma_bf16(fa[mi].v, fb[ni].v, acc[mi][ni]);
    __syncthreads();
  }

#pragma unroll
  for (int mi = 0; mi < 2; ++mi)
#pragma unroll
    for (int ni = 0; ni < 2; ++ni) {
      int gn = n0 + wn * 32 + ni * 16 + ml;
#pragma unroll
      for (int r = 0; r < 8; ++r) {
        int gm = m0 + wm * 32 + mi * 16 + hh * 8 + r;
        if (gm < M) {
          float v = acc[mi][ni][r] * out_scale;
          if (bias) v += bias[gn];
          if (ACT == 1) v = gelu_tanh(v);
          if (GATED) v = resid[(size_t)gm * N + gn] + gate[gn] * v;
          C[(size_t)gm * N + gn] = v;
        }
      }
    }
}

// ---------------------------------------------------------------------------
// Modulation GEMV: out[n] = silu(vec) . W[:,n] + b[n]
// ---------------------------------------------------------------------------
__global__ __launch_bounds__(256) void modvec_kernel(
    const float* __restrict__ vec, const float* __restrict__ W,
    const float* __restrict__ b, float* __restrict__ out, int K, int N)
{
  __shared__ float sv[3072];
  for (int i = threadIdx.x; i < K; i += 256) {
    float v = vec[i];
    sv[i] = v / (1.f + __expf(-v));
  }
  __syncthreads();
  int n = blockIdx.x * 256 + threadIdx.x;
  if (n >= N) return;
  float acc = 0.f;
  for (int k = 0; k < K; ++k) acc += sv[k] * W[(size_t)k * N + n];
  out[n] = acc + b[n];
}

// ---------------------------------------------------------------------------
// LayerNorm + modulation: out = (1+scale[n]) * ln(x_row) + shift[n]
// ---------------------------------------------------------------------------
__global__ __launch_bounds__(256) void ln_mod_kernel(
    const float* __restrict__ x, const float* __restrict__ shift,
    const float* __restrict__ scale, float* __restrict__ out)
{
  const int C = 3072;
  const int tid = threadIdx.x;
  const float* xr = x + (size_t)blockIdx.x * C;
  float s = 0.f, q = 0.f;
  for (int i = tid; i < C; i += 256) { float v = xr[i]; s += v; q += v * v; }
  __shared__ float rs[256], rq[256];
  rs[tid] = s; rq[tid] = q;
  __syncthreads();
  for (int st = 128; st > 0; st >>= 1) {
    if (tid < st) { rs[tid] += rs[tid + st]; rq[tid] += rq[tid + st]; }
    __syncthreads();
  }
  float mean = rs[0] / C;
  float var  = rq[0] / C - mean * mean;
  float inv  = rsqrtf(var + 1e-6f);
  float* orow = out + (size_t)blockIdx.x * C;
  for (int i = tid; i < C; i += 256)
    orow[i] = (1.f + scale[i]) * ((xr[i] - mean) * inv) + shift[i];
}

// ---------------------------------------------------------------------------
// QKV split + per-head RMS norm. outputs [NH][L][DH]
// ---------------------------------------------------------------------------
__global__ __launch_bounds__(128) void rms_split_kernel(
    const float* __restrict__ qkv, const float* __restrict__ qs,
    const float* __restrict__ ks, float* __restrict__ bq,
    float* __restrict__ bk, float* __restrict__ bv, int L)
{
  int l = blockIdx.x, hd = blockIdx.y, d = threadIdx.x;
  size_t base = (size_t)l * 9216 + hd * 128 + d;
  float q = qkv[base];
  float k = qkv[base + 3072];
  float v = qkv[base + 6144];
  float sq = q * q, sk = k * k;
  for (int m = 16; m >= 1; m >>= 1) { sq += __shfl_xor(sq, m); sk += __shfl_xor(sk, m); }
  __shared__ float red[2][4];
  int wv = threadIdx.x >> 5;
  if ((threadIdx.x & 31) == 0) { red[0][wv] = sq; red[1][wv] = sk; }
  __syncthreads();
  float tq = red[0][0] + red[0][1] + red[0][2] + red[0][3];
  float tk = red[1][0] + red[1][1] + red[1][2] + red[1][3];
  float fq = rsqrtf(tq / 128.f + 1e-6f);
  float fk = rsqrtf(tk / 128.f + 1e-6f);
  size_t ob = ((size_t)hd * L + l) * 128 + d;
  bq[ob] = q * fq * qs[d];
  bk[ob] = k * fk * ks[d];
  bv[ob] = v;
}

// sum over heads: out[l][d] = sum_h in[h][l][d]
__global__ __launch_bounds__(128) void hsum_kernel(
    const float* __restrict__ in, float* __restrict__ out, int L)
{
  int l = blockIdx.x, d = threadIdx.x;
  float s = 0.f;
  for (int h = 0; h < 24; ++h) s += in[((size_t)h * L + l) * 128 + d];
  out[(size_t)l * 128 + d] = s;
}

// ---------------------------------------------------------------------------
// RoPE + concat
// ---------------------------------------------------------------------------
__global__ __launch_bounds__(64) void rope_concat_kernel(
    const float* __restrict__ qa, const float* __restrict__ qb,
    const float* __restrict__ ka, const float* __restrict__ kb,
    const float* __restrict__ va, const float* __restrict__ vb,
    const float* __restrict__ pe,
    float* __restrict__ qo, float* __restrict__ ko, float* __restrict__ vo,
    int La, int Ltot)
{
  int l = blockIdx.x, hd = blockIdx.y, i = threadIdx.x;
  int Lb = Ltot - La;
  const float *qs_, *ks_, *vs_; size_t si;
  if (l < La) { qs_ = qa; ks_ = ka; vs_ = va; si = ((size_t)hd * La + l) * 128; }
  else        { qs_ = qb; ks_ = kb; vs_ = vb; si = ((size_t)hd * Lb + (l - La)) * 128; }
  float q0 = qs_[si + 2 * i], q1 = qs_[si + 2 * i + 1];
  float k0 = ks_[si + 2 * i], k1 = ks_[si + 2 * i + 1];
  float v0 = vs_[si + 2 * i], v1 = vs_[si + 2 * i + 1];
  const float* p = pe + ((size_t)l * 64 + i) * 4;
  float p00 = p[0], p01 = p[1], p10 = p[2], p11 = p[3];
  size_t oi = ((size_t)hd * Ltot + l) * 128;
  qo[oi + 2 * i]     = p00 * q0 + p01 * q1;
  qo[oi + 2 * i + 1] = p10 * q0 + p11 * q1;
  ko[oi + 2 * i]     = p00 * k0 + p01 * k1;
  ko[oi + 2 * i + 1] = p10 * k0 + p11 * k1;
  vo[oi + 2 * i]     = v0;
  vo[oi + 2 * i + 1] = v1;
}

// ---------------------------------------------------------------------------
// Flash attention (WMMA, online softmax). 128 threads = 4 waves.
// ---------------------------------------------------------------------------
__global__ __launch_bounds__(128) void flash_kernel(
    const float* __restrict__ Q, const float* __restrict__ Kt,
    const float* __restrict__ V, float* __restrict__ O,
    int Lq, int Lkv, float scale)
{
  const int head = blockIdx.y;
  const int q0   = blockIdx.x * 64;
  const int tid  = threadIdx.x;
  const int lane = tid & 31;
  const int w    = tid >> 5;
  const int ml   = lane & 15;
  const int hh   = lane >> 4;

  __shared__ __bf16 kS[64 * 136];       // K tile, row-major [kv][d]
  __shared__ __bf16 vS[128 * 72];       // V tile, transposed [d][kv]
  __shared__ __bf16 pS[4][16 * 72];     // per-wave P (16 x 64)

  const float* Qh = Q + (size_t)head * Lq * 128;
  const float* Kh = Kt + (size_t)head * Lkv * 128;
  const float* Vh = V + (size_t)head * Lkv * 128;

  Frag aQ[4];
  {
    int qrow = q0 + w * 16 + ml;
    bool qok = qrow < Lq;
    const float* qr = Qh + (size_t)(qok ? qrow : 0) * 128;
#pragma unroll
    for (int c = 0; c < 4; ++c) {
#pragma unroll
      for (int j = 0; j < 4; ++j) {
        int ka = c * 32 + hh * 8 + 2 * j;
        int kb = ka + 16;
        float a0 = qok ? qr[ka] : 0.f, a1 = qok ? qr[ka + 1] : 0.f;
        float b0 = qok ? qr[kb] : 0.f, b1 = qok ? qr[kb + 1] : 0.f;
        aQ[c].u[j]     = pack2_bf16(a0, a1);
        aQ[c].u[4 + j] = pack2_bf16(b0, b1);
      }
    }
  }

  f32x8 accO[8];
#pragma unroll
  for (int dt = 0; dt < 8; ++dt)
#pragma unroll
    for (int e = 0; e < 8; ++e) accO[dt][e] = 0.f;
  float mrun[8], lrun[8];
#pragma unroll
  for (int r = 0; r < 8; ++r) { mrun[r] = -1e30f; lrun[r] = 0.f; }

  for (int kv0 = 0; kv0 < Lkv; kv0 += 64) {
    for (int i = 0; i < 64; ++i) {
      int e  = tid + i * 128;
      int rr = e >> 7, cc = e & 127;
      int kr = kv0 + rr;
      float kf = (kr < Lkv) ? Kh[(size_t)kr * 128 + cc] : 0.f;
      float vf = (kr < Lkv) ? Vh[(size_t)kr * 128 + cc] : 0.f;
      kS[rr * 136 + cc] = to_bf16(kf);
      vS[cc * 72 + rr]  = to_bf16(vf);
    }
    __syncthreads();

    float pv[4][8];
#pragma unroll
    for (int nt = 0; nt < 4; ++nt) {
      f32x8 s;
#pragma unroll
      for (int e = 0; e < 8; ++e) s[e] = 0.f;
      int col = nt * 16 + ml;
#pragma unroll
      for (int c = 0; c < 4; ++c) {
        Frag bK;
#pragma unroll
        for (int j = 0; j < 8; ++j)
          bK.u[j] = *reinterpret_cast<const unsigned*>(&kS[col * 136 + c * 32 + (j >> 2) * 16 + hh * 8 + 2 * (j & 3)]);
        s = wmma_bf16(aQ[c].v, bK.v, s);
      }
      bool colok = (kv0 + col) < Lkv;
#pragma unroll
      for (int r = 0; r < 8; ++r)
        pv[nt][r] = colok ? s[r] * scale : -1e30f;
    }

#pragma unroll
    for (int r = 0; r < 8; ++r) {
      float mx = fmaxf(fmaxf(pv[0][r], pv[1][r]), fmaxf(pv[2][r], pv[3][r]));
      for (int m = 8; m >= 1; m >>= 1) mx = fmaxf(mx, __shfl_xor(mx, m));
      float mnew = fmaxf(mrun[r], mx);
      float corr = __expf(mrun[r] - mnew);
      mrun[r] = mnew;
      float rowsum = 0.f;
#pragma unroll
      for (int nt = 0; nt < 4; ++nt) {
        float p = __expf(pv[nt][r] - mnew);
        pv[nt][r] = p;
        rowsum += p;
      }
      for (int m = 8; m >= 1; m >>= 1) rowsum += __shfl_xor(rowsum, m);
      lrun[r] = lrun[r] * corr + rowsum;
#pragma unroll
      for (int dt = 0; dt < 8; ++dt) accO[dt][r] *= corr;
      int prow = hh * 8 + r;
#pragma unroll
      for (int nt = 0; nt < 4; ++nt)
        pS[w][prow * 72 + nt * 16 + ml] = to_bf16(pv[nt][r]);
    }
    asm volatile("s_wait_dscnt 0" ::: "memory");

#pragma unroll
    for (int kc = 0; kc < 2; ++kc) {
      Frag aP;
#pragma unroll
      for (int j = 0; j < 4; ++j) {
        aP.u[j]     = *reinterpret_cast<const unsigned*>(&pS[w][ml * 72 + kc * 32 + hh * 8 + 2 * j]);
        aP.u[4 + j] = *reinterpret_cast<const unsigned*>(&pS[w][ml * 72 + kc * 32 + 16 + hh * 8 + 2 * j]);
      }
#pragma unroll
      for (int dt = 0; dt < 8; ++dt) {
        Frag bV;
        int dcol = dt * 16 + ml;
#pragma unroll
        for (int j = 0; j < 8; ++j)
          bV.u[j] = *reinterpret_cast<const unsigned*>(&vS[dcol * 72 + kc * 32 + (j >> 2) * 16 + hh * 8 + 2 * (j & 3)]);
        accO[dt] = wmma_bf16(aP.v, bV.v, accO[dt]);
      }
    }
    __syncthreads();
  }

  float* Oh = O + (size_t)head * Lq * 128;
#pragma unroll
  for (int r = 0; r < 8; ++r) {
    int row = q0 + w * 16 + hh * 8 + r;
    if (row < Lq) {
      float invl = (lrun[r] > 0.f) ? 1.f / lrun[r] : 0.f;
#pragma unroll
      for (int dt = 0; dt < 8; ++dt)
        Oh[(size_t)row * 128 + dt * 16 + ml] = accO[dt][r] * invl;
    }
  }
}

// merge heads: out[l][h*128+d] = attn[h][off+l][d]
__global__ __launch_bounds__(128) void merge_kernel(
    const float* __restrict__ attn, float* __restrict__ out,
    int Lsub, int off, int Ltot)
{
  int l = blockIdx.x, hd = blockIdx.y, d = threadIdx.x;
  out[(size_t)l * 3072 + hd * 128 + d] = attn[((size_t)hd * Ltot + off + l) * 128 + d];
}

// ---------------------------------------------------------------------------
// Host-side orchestration
// ---------------------------------------------------------------------------
extern "C" void kernel_launch(void* const* d_in, const int* in_sizes, int n_in,
                              void* d_out, int out_size, void* d_ws, size_t ws_size,
                              hipStream_t stream) {
  (void)in_sizes; (void)n_in; (void)out_size;

  const float* img    = (const float*)d_in[0];
  const float* txt    = (const float*)d_in[1];
  const float* vec    = (const float*)d_in[2];
  const float* pe     = (const float*)d_in[3];
  const float* con    = (const float*)d_in[4];
  const float* cvec   = (const float*)d_in[5];
  const float* cpe    = (const float*)d_in[6];
  const float* imodw  = (const float*)d_in[7];
  const float* imodb  = (const float*)d_in[8];
  const float* tmodw  = (const float*)d_in[9];
  const float* tmodb  = (const float*)d_in[10];
  const float* iqkvw  = (const float*)d_in[11];
  const float* iqkvb  = (const float*)d_in[12];
  const float* iqs    = (const float*)d_in[13];
  const float* iks    = (const float*)d_in[14];
  const float* iprojw = (const float*)d_in[15];
  const float* iprojb = (const float*)d_in[16];
  const float* imlp1w = (const float*)d_in[17];
  const float* imlp1b = (const float*)d_in[18];
  const float* imlp2w = (const float*)d_in[19];
  const float* imlp2b = (const float*)d_in[20];
  const float* tqkvw  = (const float*)d_in[21];
  const float* tqkvb  = (const float*)d_in[22];
  const float* tqs    = (const float*)d_in[23];
  const float* tks    = (const float*)d_in[24];
  const float* tprojw = (const float*)d_in[25];
  const float* tprojb = (const float*)d_in[26];
  const float* tmlp1w = (const float*)d_in[27];
  const float* tmlp1b = (const float*)d_in[28];
  const float* tmlp2w = (const float*)d_in[29];
  const float* tmlp2b = (const float*)d_in[30];

  const int HID = 3072, NH = 24, DH = 128, MLPD = 12288;
  const int LI = 1024, LT = 512, LC = 20;
  const int L1 = LT + LI;   // 1536
  const int L2 = LC + LI;   // 1044
  const float sm_scale = 0.08838834764831845f; // 1/sqrt(128)

  // ---- workspace layout (floats; allocs rounded to 32B for b128 DMA) ----
  float* W = (float*)d_ws;
  size_t off = 0;
  auto alloc = [&](size_t n) {
    float* p = W + off;
    off += (n + 7) & ~(size_t)7;
    return p;
  };

  float* mod_i = alloc(6 * HID);
  float* mod_t = alloc(6 * HID);
  float* mod_c = alloc(6 * HID);
  float* xm_i  = alloc((size_t)LI * HID);
  float* xm_t  = alloc((size_t)LT * HID);
  float* xm_c  = alloc((size_t)LC * HID);
  float* qkv_i = alloc((size_t)LI * 3 * HID); // reused: q1 | k1
  float* qkv_t = alloc((size_t)LT * 3 * HID); // reused: v1
  float* qkv_c = alloc((size_t)LC * 3 * HID);
  float* bq_i = alloc((size_t)NH * LI * DH);
  float* bk_i = alloc((size_t)NH * LI * DH);
  float* bv_i = alloc((size_t)NH * LI * DH);
  float* bq_t = alloc((size_t)NH * LT * DH);
  float* bk_t = alloc((size_t)NH * LT * DH);
  float* bv_t = alloc((size_t)NH * LT * DH);
  float* bq_c = alloc((size_t)NH * LC * DH);
  float* bk_c = alloc((size_t)NH * LC * DH);
  float* bv_c = alloc((size_t)NH * LC * DH);
  float* hq_i = alloc((size_t)LI * DH);
  float* hq_c = alloc((size_t)LC * DH);
  float* q2 = alloc((size_t)NH * L2 * DH);
  float* k2 = alloc((size_t)NH * L2 * DH);
  float* v2 = alloc((size_t)NH * L2 * DH);
  float* attn1 = alloc((size_t)NH * L1 * DH);
  float* attn2 = alloc((size_t)NH * LC * DH);
  float* txtam = alloc((size_t)LT * HID);
  float* imgam = alloc((size_t)LI * HID);
  float* conam = alloc((size_t)LC * HID);
  float* img1 = alloc((size_t)LI * HID);
  float* txt1 = alloc((size_t)LT * HID);
  float* con1 = alloc((size_t)LC * HID);
  float* hbuf = alloc((size_t)LI * MLPD);

  // bf16 transposed weight copies (optional fast path)
  auto alloc_bf = [&](size_t nelem) { return (__bf16*)alloc((nelem + 1) / 2); };
  __bf16* iqkvT  = alloc_bf((size_t)HID * 3 * HID);
  __bf16* tqkvT  = alloc_bf((size_t)HID * 3 * HID);
  __bf16* iprojT = alloc_bf((size_t)HID * HID);
  __bf16* tprojT = alloc_bf((size_t)HID * HID);
  __bf16* imlp1T = alloc_bf((size_t)HID * MLPD);
  __bf16* tmlp1T = alloc_bf((size_t)HID * MLPD);
  __bf16* imlp2T = alloc_bf((size_t)MLPD * HID);
  __bf16* tmlp2T = alloc_bf((size_t)MLPD * HID);
  const bool fastw = (off * sizeof(float)) <= ws_size;

  float* q1 = qkv_i;
  float* k1 = qkv_i + (size_t)NH * L1 * DH;
  float* v1 = qkv_t;

  float* out_img  = (float*)d_out;
  float* out_txt  = out_img + (size_t)LI * HID;
  float* out_con  = out_txt + (size_t)LT * HID;
  float* out_cam  = out_con + (size_t)LC * HID;
  float* out_cmap = out_cam + (size_t)LC * LI;

  // 0) weight conversion pre-pass (fast path)
  if (fastw) {
    convT_kernel<<<dim3(3 * HID / 32, HID / 32), 256, 0, stream>>>(iqkvw, iqkvT, HID, 3 * HID);
    convT_kernel<<<dim3(3 * HID / 32, HID / 32), 256, 0, stream>>>(tqkvw, tqkvT, HID, 3 * HID);
    convT_kernel<<<dim3(HID / 32, HID / 32), 256, 0, stream>>>(iprojw, iprojT, HID, HID);
    convT_kernel<<<dim3(HID / 32, HID / 32), 256, 0, stream>>>(tprojw, tprojT, HID, HID);
    convT_kernel<<<dim3(MLPD / 32, HID / 32), 256, 0, stream>>>(imlp1w, imlp1T, HID, MLPD);
    convT_kernel<<<dim3(MLPD / 32, HID / 32), 256, 0, stream>>>(tmlp1w, tmlp1T, HID, MLPD);
    convT_kernel<<<dim3(HID / 32, MLPD / 32), 256, 0, stream>>>(imlp2w, imlp2T, MLPD, HID);
    convT_kernel<<<dim3(HID / 32, MLPD / 32), 256, 0, stream>>>(tmlp2w, tmlp2T, MLPD, HID);
  }

  // 1) modulation GEMVs
  modvec_kernel<<<72, 256, 0, stream>>>(vec,  imodw, imodb, mod_i, HID, 6 * HID);
  modvec_kernel<<<72, 256, 0, stream>>>(vec,  tmodw, tmodb, mod_t, HID, 6 * HID);
  modvec_kernel<<<72, 256, 0, stream>>>(cvec, tmodw, tmodb, mod_c, HID, 6 * HID);

  // 2) LN + modulate (chunks: s1,sc1,g1,s2,sc2,g2)
  ln_mod_kernel<<<LI, 256, 0, stream>>>(img, mod_i, mod_i + HID, xm_i);
  ln_mod_kernel<<<LT, 256, 0, stream>>>(txt, mod_t, mod_t + HID, xm_t);
  ln_mod_kernel<<<LC, 256, 0, stream>>>(con, mod_c, mod_c + HID, xm_c);

  // 3) QKV projections
  if (fastw) {
    gemm_bf16t_kernel<0,0><<<dim3(144, 8), 256, 0, stream>>>(xm_i, iqkvT, iqkvb, nullptr, nullptr, qkv_i, LI, 3*HID, HID, 1.f);
    gemm_bf16t_kernel<0,0><<<dim3(144, 4), 256, 0, stream>>>(xm_t, tqkvT, tqkvb, nullptr, nullptr, qkv_t, LT, 3*HID, HID, 1.f);
    gemm_bf16t_kernel<0,0><<<dim3(144, 1), 256, 0, stream>>>(xm_c, tqkvT, tqkvb, nullptr, nullptr, qkv_c, LC, 3*HID, HID, 1.f);
  } else {
    gemm_kernel<0,0,0><<<dim3(144, 8), 256, 0, stream>>>(xm_i, iqkvw, iqkvb, nullptr, nullptr, qkv_i, LI, 3*HID, HID, 1.f);
    gemm_kernel<0,0,0><<<dim3(144, 4), 256, 0, stream>>>(xm_t, tqkvw, tqkvb, nullptr, nullptr, qkv_t, LT, 3*HID, HID, 1.f);
    gemm_kernel<0,0,0><<<dim3(144, 1), 256, 0, stream>>>(xm_c, tqkvw, tqkvb, nullptr, nullptr, qkv_c, LC, 3*HID, HID, 1.f);
  }

  // 4) split + RMS norm
  rms_split_kernel<<<dim3(LI, NH), 128, 0, stream>>>(qkv_i, iqs, iks, bq_i, bk_i, bv_i, LI);
  rms_split_kernel<<<dim3(LT, NH), 128, 0, stream>>>(qkv_t, tqs, tks, bq_t, bk_t, bv_t, LT);
  rms_split_kernel<<<dim3(LC, NH), 128, 0, stream>>>(qkv_c, tqs, tks, bq_c, bk_c, bv_c, LC);

  // 5) head sums (pre-rope q) for cross-attention maps
  hsum_kernel<<<LI, 128, 0, stream>>>(bq_i, hq_i, LI);
  hsum_kernel<<<LC, 128, 0, stream>>>(bq_c, hq_c, LC);

  // 6) RoPE + concat (overwrites qkv buffers -> q1/k1/v1)
  rope_concat_kernel<<<dim3(L1, NH), 64, 0, stream>>>(bq_t, bq_i, bk_t, bk_i, bv_t, bv_i, pe,  q1, k1, v1, LT, L1);
  rope_concat_kernel<<<dim3(L2, NH), 64, 0, stream>>>(bq_c, bq_i, bk_c, bk_i, bv_c, bv_i, cpe, q2, k2, v2, LC, L2);

  // 7) flash attention (attn2 only needs its first 20 query rows)
  flash_kernel<<<dim3(L1 / 64, NH), 128, 0, stream>>>(q1, k1, v1, attn1, L1, L1, sm_scale);
  flash_kernel<<<dim3(1, NH),        128, 0, stream>>>(q2, k2, v2, attn2, LC, L2, sm_scale);

  // 8) merge heads
  merge_kernel<<<dim3(LT, NH), 128, 0, stream>>>(attn1, txtam, LT, 0,  L1);
  merge_kernel<<<dim3(LI, NH), 128, 0, stream>>>(attn1, imgam, LI, LT, L1);
  merge_kernel<<<dim3(LC, NH), 128, 0, stream>>>(attn2, conam, LC, 0,  LC);

  // 9) cross_attention_maps = (sum_h cq) (sum_h iq)^T / NH
  gemm_kernel<1,0,0><<<dim3(16, 1), 256, 0, stream>>>(hq_c, hq_i, nullptr, nullptr, nullptr, out_cam, LC, LI, DH, 1.f / NH);
  // 10) concept_attention_maps = conam imgam^T
  gemm_kernel<1,0,0><<<dim3(16, 1), 256, 0, stream>>>(conam, imgam, nullptr, nullptr, nullptr, out_cmap, LC, LI, HID, 1.f);

  // 11) attention projection + gated residual
  if (fastw) {
    gemm_bf16t_kernel<0,1><<<dim3(48, 8), 256, 0, stream>>>(imgam, iprojT, iprojb, mod_i + 2*HID, img, img1, LI, HID, HID, 1.f);
    gemm_bf16t_kernel<0,1><<<dim3(48, 4), 256, 0, stream>>>(txtam, tprojT, tprojb, mod_t + 2*HID, txt, txt1, LT, HID, HID, 1.f);
    gemm_bf16t_kernel<0,1><<<dim3(48, 1), 256, 0, stream>>>(conam, tprojT, tprojb, mod_c + 2*HID, con, con1, LC, HID, HID, 1.f);
  } else {
    gemm_kernel<0,0,1><<<dim3(48, 8), 256, 0, stream>>>(imgam, iprojw, iprojb, mod_i + 2*HID, img, img1, LI, HID, HID, 1.f);
    gemm_kernel<0,0,1><<<dim3(48, 4), 256, 0, stream>>>(txtam, tprojw, tprojb, mod_t + 2*HID, txt, txt1, LT, HID, HID, 1.f);
    gemm_kernel<0,0,1><<<dim3(48, 1), 256, 0, stream>>>(conam, tprojw, tprojb, mod_c + 2*HID, con, con1, LC, HID, HID, 1.f);
  }

  // 12) MLP per stream (serialized on stream -> shared hidden buffer)
  ln_mod_kernel<<<LI, 256, 0, stream>>>(img1, mod_i + 3*HID, mod_i + 4*HID, xm_i);
  if (fastw) {
    gemm_bf16t_kernel<1,0><<<dim3(192, 8), 256, 0, stream>>>(xm_i, imlp1T, imlp1b, nullptr, nullptr, hbuf, LI, MLPD, HID, 1.f);
    gemm_bf16t_kernel<0,1><<<dim3(48, 8),  256, 0, stream>>>(hbuf, imlp2T, imlp2b, mod_i + 5*HID, img1, out_img, LI, HID, MLPD, 1.f);
  } else {
    gemm_kernel<0,1,0><<<dim3(192, 8), 256, 0, stream>>>(xm_i, imlp1w, imlp1b, nullptr, nullptr, hbuf, LI, MLPD, HID, 1.f);
    gemm_kernel<0,0,1><<<dim3(48, 8),  256, 0, stream>>>(hbuf, imlp2w, imlp2b, mod_i + 5*HID, img1, out_img, LI, HID, MLPD, 1.f);
  }

  ln_mod_kernel<<<LT, 256, 0, stream>>>(txt1, mod_t + 3*HID, mod_t + 4*HID, xm_t);
  if (fastw) {
    gemm_bf16t_kernel<1,0><<<dim3(192, 4), 256, 0, stream>>>(xm_t, tmlp1T, tmlp1b, nullptr, nullptr, hbuf, LT, MLPD, HID, 1.f);
    gemm_bf16t_kernel<0,1><<<dim3(48, 4),  256, 0, stream>>>(hbuf, tmlp2T, tmlp2b, mod_t + 5*HID, txt1, out_txt, LT, HID, MLPD, 1.f);
  } else {
    gemm_kernel<0,1,0><<<dim3(192, 4), 256, 0, stream>>>(xm_t, tmlp1w, tmlp1b, nullptr, nullptr, hbuf, LT, MLPD, HID, 1.f);
    gemm_kernel<0,0,1><<<dim3(48, 4),  256, 0, stream>>>(hbuf, tmlp2w, tmlp2b, mod_t + 5*HID, txt1, out_txt, LT, HID, MLPD, 1.f);
  }

  ln_mod_kernel<<<LC, 256, 0, stream>>>(con1, mod_c + 3*HID, mod_c + 4*HID, xm_c);
  if (fastw) {
    gemm_bf16t_kernel<1,0><<<dim3(192, 1), 256, 0, stream>>>(xm_c, tmlp1T, tmlp1b, nullptr, nullptr, hbuf, LC, MLPD, HID, 1.f);
    gemm_bf16t_kernel<0,1><<<dim3(48, 1),  256, 0, stream>>>(hbuf, tmlp2T, tmlp2b, mod_c + 5*HID, con1, out_con, LC, HID, MLPD, 1.f);
  } else {
    gemm_kernel<0,1,0><<<dim3(192, 1), 256, 0, stream>>>(xm_c, tmlp1w, tmlp1b, nullptr, nullptr, hbuf, LC, MLPD, HID, 1.f);
    gemm_kernel<0,0,1><<<dim3(48, 1),  256, 0, stream>>>(hbuf, tmlp2w, tmlp2b, mod_c + 5*HID, con1, out_con, LC, HID, MLPD, 1.f);
  }
}
